// DualBranchSTN_9397388443891
// MI455X (gfx1250) — matrix operand order repaired and checked by reading, not compiled, not run on hardware
//
#include <hip/hip_runtime.h>
#include <math.h>

typedef unsigned short u16;
typedef unsigned int   u32;
typedef long long      i64;

#define Bn 32
#define Tt 12
#define Nn 64
#define Ff 64
#define Hh 256
#define H2 128
#define Mm 768
#define Rr (Bn*Mm)          /* 24576 */
#define BN_EPS 1e-5f

typedef __attribute__((ext_vector_type(16))) __bf16 v16bf;
typedef __attribute__((ext_vector_type(8)))  float  v8f;

__device__ __forceinline__ u16 f2bf(float f){
  u32 u = __float_as_uint(f);
  u32 r = (u + 0x7FFFu + ((u >> 16) & 1u)) >> 16;   // RNE truncate f32 -> bf16
  return (u16)r;
}
__device__ __forceinline__ float sigm(float x){ return 1.f/(1.f+expf(-x)); }

// CDNA5 async global->LDS copy (ASYNCcnt-tracked), 16 bytes per lane.
__device__ __forceinline__ void async_b128(u16* lds, const u16* g){
  asm volatile("global_load_async_to_lds_b128 %0, %1, off"
               :: "v"((u32)(uintptr_t)lds),
                  "v"((unsigned long long)(uintptr_t)g)
               : "memory");
}
__device__ __forceinline__ void wait_async0(){
  asm volatile("s_wait_asynccnt 0x0" ::: "memory");
}

// ---------------------------------------------------------------------------
// bf16 WMMA GEMM: C = A*B [+ A2*B2] + bias
//  * 128 threads = 4 waves; block computes 64x64 C tile; wave does 32x32
//  * A: [rows,K] row-major bf16 (lda); B: [N,K] row-major bf16 (ldb) — all
//    operands pre-transposed so both tiles are contiguous async copies
//  * double-buffered LDS; tile k+1 streamed with global_load_async_to_lds_b128
//    while WMMAs consume tile k; s_wait_asynccnt + barrier between stages
//  * batched via blockIdx.z (stride 0 = broadcast, used for physics L)
//  * fragment packing per CDNA5 ISA 7.12.2 (wave32 16x16x32 bf16 layouts)
// ---------------------------------------------------------------------------
union FragAB { uint4 u[2]; v16bf v; };
union FragC  { v8f v; float f[8]; };

__global__ __launch_bounds__(128) void gemm_bf16_kernel(
    const u16* __restrict__ A,  i64 sAb, int lda,
    const u16* __restrict__ Bm, i64 sBb, int ldb,
    const u16* __restrict__ A2, int lda2,
    const u16* __restrict__ B2, int ldb2,
    int nSrc,
    const float* __restrict__ bias,
    float* __restrict__ Cf, u16* __restrict__ Cbf, i64 sCb, int ldc,
    int Kdim)
{
  __shared__ __align__(16) u16 As[2][64*32];
  __shared__ __align__(16) u16 Bs[2][64*32];
  const int tid   = threadIdx.x;
  const int lane  = tid & 31;
  const int wv    = tid >> 5;
  const int lrow  = lane & 15;
  const int lhalf = lane >> 4;
  const int wr = (wv & 1) * 32, wc = (wv >> 1) * 32;
  const i64 bz = blockIdx.z;
  const int row0 = blockIdx.x * 64, col0 = blockIdx.y * 64;
  const int r = tid >> 1, cp = (tid & 1) * 16;   // staging: 16 bf16 per thread

  FragC acc[2][2];
#pragma unroll
  for (int i = 0; i < 2; ++i)
#pragma unroll
    for (int j = 0; j < 2; ++j)
#pragma unroll
      for (int e = 0; e < 8; ++e) acc[i][j].f[e] = 0.f;

  for (int src = 0; src < nSrc; ++src) {
    const u16* Ab = src ? A2 : (A  + bz * sAb);
    const u16* Bb = src ? B2 : (Bm + bz * sBb);
    const int  la = src ? lda2 : lda;
    const int  lb = src ? ldb2 : ldb;
    const i64  aRow = (i64)(row0 + r) * la + cp;
    const i64  bRow = (i64)(col0 + r) * lb + cp;
    const int  nk = Kdim >> 5;

    // preload tile 0
    async_b128(&As[0][r * 32 + cp], Ab + aRow);
    async_b128(&Bs[0][r * 32 + cp], Bb + bRow);

    for (int t = 0; t < nk; ++t) {
      wait_async0();        // this wave's async copies have landed in LDS
      __syncthreads();      // all waves done reading the other buffer
      const int cur = t & 1;
      if (t + 1 < nk) {     // stream next tile while computing this one
        const int kc = (t + 1) * 32;
        async_b128(&As[cur ^ 1][r * 32 + cp], Ab + aRow + kc);
        async_b128(&Bs[cur ^ 1][r * 32 + cp], Bb + bRow + kc);
      }
#pragma unroll
      for (int i = 0; i < 2; ++i) {
        FragAB a;  // A frag: row lrow, K = {0..7,16..23} + 8*lhalf
        const uint4* ap = reinterpret_cast<const uint4*>(&As[cur][(wr + i * 16 + lrow) * 32]);
        a.u[0] = ap[lhalf]; a.u[1] = ap[2 + lhalf];
#pragma unroll
        for (int j = 0; j < 2; ++j) {
          FragAB bfr; // B frag: col lrow, K = {0..7,16..23} + 8*lhalf
          const uint4* bp = reinterpret_cast<const uint4*>(&Bs[cur][(wc + j * 16 + lrow) * 32]);
          bfr.u[0] = bp[lhalf]; bfr.u[1] = bp[2 + lhalf];
          acc[i][j].v = __builtin_amdgcn_wmma_f32_16x16x32_bf16(
              false, a.v, false, bfr.v, (short)0, acc[i][j].v, false, false);
        }
      }
    }
    __syncthreads();        // protect buffers before next source's preload
  }

  // epilogue: C/D layout — VGPR e, lanes 0-15: M=e, N=lane; lanes 16-31: M=e+8
#pragma unroll
  for (int i = 0; i < 2; ++i)
#pragma unroll
    for (int j = 0; j < 2; ++j)
#pragma unroll
      for (int e = 0; e < 8; ++e) {
        const int gr = row0 + wr + i * 16 + lhalf * 8 + e;
        const int gc = col0 + wc + j * 16 + lrow;
        float v = acc[i][j].f[e];
        if (bias) v += bias[gc];
        const i64 o = bz * sCb + (i64)gr * ldc + gc;
        if (Cf)  Cf[o]  = v;
        if (Cbf) Cbf[o] = f2bf(v);
      }
}

// ---------------------------------------------------------------------------
// Small support kernels
// ---------------------------------------------------------------------------
__global__ void cvt_kernel(const float* __restrict__ s, u16* __restrict__ d, int n){
  int i = blockIdx.x * blockDim.x + threadIdx.x;
  if (i < n) d[i] = f2bf(s[i]);
}

// f32 [rows,cols] -> bf16 [cols,rows] (transpose-convert, for w0/w1)
__global__ void cvtT_kernel(const float* __restrict__ s, u16* __restrict__ d,
                            int rows, int cols){
  int i = blockIdx.x * blockDim.x + threadIdx.x;
  if (i >= rows * cols) return;
  int rr = i / cols, cc = i % cols;
  d[(i64)cc * rows + rr] = f2bf(s[i]);
}

// h0 = x + temp_embed + spat_embed  (f32 + bf16 + transposed-bf16)
__global__ void prep_kernel(const float* __restrict__ x, const float* __restrict__ te,
                            const float* __restrict__ se, float* __restrict__ Hf,
                            u16* __restrict__ Hbf, u16* __restrict__ HbfT){
  int gid = blockIdx.x * blockDim.x + threadIdx.x;
  int f = gid & 63, row = gid >> 6;
  int m = row % Mm, b = row / Mm, t = m / Nn, n = m % Nn;
  float v = x[gid] + te[t * Ff + f] + se[n * Ff + f];
  Hf[gid] = v;
  u16 bf = f2bf(v);
  Hbf[gid] = bf;
  HbfT[((i64)b * Ff + f) * Mm + m] = bf;   // [B][C][M]
}

// cosine-normalized fun_x rows -> bf16
__global__ void xn_kernel(const float* __restrict__ fx, u16* __restrict__ xnbf){
  __shared__ float sm[64];
  int row = blockIdx.x, f = threadIdx.x;
  float v = fx[(i64)row * Ff + f];
  sm[f] = v * v; __syncthreads();
  for (int s = 32; s > 0; s >>= 1){ if (f < s) sm[f] += sm[f + s]; __syncthreads(); }
  float inv = 1.f / fmaxf(sqrtf(sm[0]), 1e-12f);
  xnbf[(i64)row * Ff + f] = f2bf(v * inv);
}

__global__ void zdiag_kernel(float* __restrict__ S){
  int b = blockIdx.x, i = threadIdx.x;
  S[(i64)b * Mm * Mm + (i64)i * Mm + i] = 0.f;
}

// per-row top-8 -> membership bitmask (24 u32 per row)
__global__ void topk_kernel(const float* __restrict__ S, u32* __restrict__ bm){
  int gid = blockIdx.x * blockDim.x + threadIdx.x;
  if (gid >= Rr) return;
  const float* row = S + (i64)gid * Mm;          // (b*M+i)*M == b*M*M + i*M
  float v[8]; int id[8];
#pragma unroll
  for (int t = 0; t < 8; ++t){ v[t] = -3.4e38f; id[t] = -1; }
  for (int j = 0; j < Mm; ++j){
    float x = row[j];
    if (x > v[7]){
      int p = 7;
      while (p > 0 && x > v[p-1]){ v[p] = v[p-1]; id[p] = id[p-1]; --p; }
      v[p] = x; id[p] = j;
    }
  }
  u32 w[24];
  for (int k = 0; k < 24; ++k) w[k] = 0u;
  for (int t = 0; t < 8; ++t) if (id[t] >= 0) w[id[t] >> 5] |= (1u << (id[t] & 31));
  u32* o = bm + (i64)gid * 24;
  for (int k = 0; k < 24; ++k) o[k] = w[k];
}

// mask_ij = (S_ij>0) && (j in topk_i || i in topk_j);  dis = deg^-1/2
__global__ void deg_kernel(const float* __restrict__ S, const u32* __restrict__ bm,
                           float* __restrict__ dis){
  __shared__ int red[64];
  int gid = blockIdx.x;
  int b = gid / Mm, i = gid % Mm;
  const float* row = S + (i64)gid * Mm;
  const u32* bmi = bm + (i64)gid * 24;
  int cnt = 0;
  for (int j = threadIdx.x; j < Mm; j += 64){
    float s = row[j];
    if (s > 0.f){
      bool ti = (bmi[j >> 5] >> (j & 31)) & 1u;
      bool tj = (bm[((i64)b * Mm + j) * 24 + (i >> 5)] >> (i & 31)) & 1u;
      if (ti || tj) ++cnt;
    }
  }
  red[threadIdx.x] = cnt; __syncthreads();
  for (int s = 32; s > 0; s >>= 1){ if (threadIdx.x < s) red[threadIdx.x] += red[threadIdx.x + s]; __syncthreads(); }
  if (threadIdx.x == 0){
    int d = red[0];
    dis[gid] = d > 0 ? rsqrtf((float)d) : 0.f;
  }
}

__global__ void ldyn_kernel(const float* __restrict__ S, const u32* __restrict__ bm,
                            const float* __restrict__ dis, u16* __restrict__ L){
  int gid = blockIdx.x;
  int b = gid / Mm, i = gid % Mm;
  const float* row = S + (i64)gid * Mm;
  const u32* bmi = bm + (i64)gid * 24;
  float di = dis[gid];
  for (int j = threadIdx.x; j < Mm; j += 64){
    float s = row[j];
    u16 o = 0;
    if (s > 0.f){
      bool ti = (bmi[j >> 5] >> (j & 31)) & 1u;
      bool tj = (bm[((i64)b * Mm + j) * 24 + (i >> 5)] >> (i & 31)) & 1u;
      if (ti || tj) o = f2bf(-di * dis[(i64)b * Mm + j]);
    }
    L[(i64)gid * Mm + j] = o;
  }
}

__global__ void lphy_kernel(u16* __restrict__ L){
  int i = blockIdx.x;
  int ti = i / Nn;
  float di = rsqrtf((float)(Nn * (1 + (ti > 0) + (ti < Tt - 1))));
  for (int j = threadIdx.x; j < Mm; j += blockDim.x){
    int tj = j / Nn;
    int d = ti - tj; if (d < 0) d = -d;
    u16 o = 0;
    if (d <= 1){
      float dj = rsqrtf((float)(Nn * (1 + (tj > 0) + (tj < Tt - 1))));
      o = f2bf(-di * dj);
    }
    L[(i64)i * Mm + j] = o;
  }
}

// BatchNorm training stats (biased var) over all B*M rows
__global__ void bnstats_kernel(const float* __restrict__ Y, int cout,
                               float* __restrict__ mu, float* __restrict__ rs){
  __shared__ float r1[256], r2[256];
  int c = blockIdx.x;
  float s = 0.f, s2 = 0.f;
  for (int r = threadIdx.x; r < Rr; r += 256){
    float v = Y[(i64)r * cout + c];
    s += v; s2 += v * v;
  }
  r1[threadIdx.x] = s; r2[threadIdx.x] = s2; __syncthreads();
  for (int k = 128; k > 0; k >>= 1){
    if (threadIdx.x < k){ r1[threadIdx.x] += r1[threadIdx.x + k]; r2[threadIdx.x] += r2[threadIdx.x + k]; }
    __syncthreads();
  }
  if (threadIdx.x == 0){
    float m = r1[0] / (float)Rr;
    float v = fmaxf(r2[0] / (float)Rr - m * m, 0.f);
    mu[c] = m; rs[c] = rsqrtf(v + BN_EPS);
  }
}

// h = relu(BN(y) + res)  -> f32 + bf16 + transposed-bf16
__global__ void bnfin_kernel(const float* __restrict__ Y, const float* __restrict__ RES,
                             const float* __restrict__ g, const float* __restrict__ be,
                             const float* __restrict__ mu, const float* __restrict__ rs,
                             float* __restrict__ Hf, u16* __restrict__ Hbf,
                             u16* __restrict__ HbfT, int cout){
  i64 gid = (i64)blockIdx.x * blockDim.x + threadIdx.x;
  int c = (int)(gid % cout);
  i64 row = gid / cout;
  int b = (int)(row / Mm), m = (int)(row % Mm);
  float y = g[c] * (Y[gid] - mu[c]) * rs[c] + be[c] + RES[gid];
  y = fmaxf(y, 0.f);
  Hf[gid] = y;
  u16 bf = f2bf(y);
  Hbf[gid] = bf;
  HbfT[((i64)b * cout + c) * Mm + m] = bf;
}

// fusion: per-node channel-attention MLP + per-timestep gating head -> alpha
__global__ __launch_bounds__(128) void alpha_kernel(
    const float* __restrict__ xphy, const float* __restrict__ xfun,
    const float* __restrict__ cw1, const float* __restrict__ cb1,
    const float* __restrict__ cw2, const float* __restrict__ cb2,
    const float* __restrict__ tw1, const float* __restrict__ tb1,
    const float* __restrict__ tw2, const float* __restrict__ tb2,
    float* __restrict__ alpha){
  __shared__ float s[128], u[64], fu[128], hg[64], red[64];
  int row = blockIdx.x, tid = threadIdx.x;
  int t = (row % Mm) / Nn;
  s[tid] = (tid < 64) ? xphy[(i64)row * 64 + tid] : xfun[(i64)row * 64 + tid - 64];
  __syncthreads();
  if (tid < 64){
    float a = cb1[tid];
    for (int k = 0; k < 128; ++k) a += s[k] * cw1[tid * 128 + k];
    u[tid] = fmaxf(a, 0.f);
  }
  __syncthreads();
  {
    float a = cb2[tid];
    for (int j = 0; j < 64; ++j) a += u[j] * cw2[tid * 64 + j];
    fu[tid] = s[tid] * sigm(a);
  }
  __syncthreads();
  if (tid < 64){
    float a = tb1[t * 64 + tid];
    const float* w = tw1 + ((i64)t * 64 + tid) * 128;
    for (int c = 0; c < 128; ++c) a += fu[c] * w[c];
    hg[tid] = fmaxf(a, 0.f);
  }
  __syncthreads();
  if (tid < 64) red[tid] = hg[tid] * tw2[t * 64 + tid];
  __syncthreads();
  for (int k = 32; k > 0; k >>= 1){ if (tid < k) red[tid] += red[tid + k]; __syncthreads(); }
  if (tid == 0) alpha[row] = sigm(red[0] + tb2[t]);
}

// g_phy = mean(alpha * x_phy), g_fun = mean((1-alpha) * x_fun) over T,N
__global__ void gpool_kernel(const float* __restrict__ alpha,
                             const float* __restrict__ xphy, const float* __restrict__ xfun,
                             float* __restrict__ gph, float* __restrict__ gfn){
  int b = blockIdx.x, f = threadIdx.x;
  float ap = 0.f, af = 0.f;
  for (int m = 0; m < Mm; ++m){
    float al = alpha[b * Mm + m];
    i64 o = ((i64)b * Mm + m) * 64 + f;
    ap += al * xphy[o];
    af += (1.f - al) * xfun[o];
  }
  gph[b * 64 + f] = ap * (1.f / 768.f);
  gfn[b * 64 + f] = af * (1.f / 768.f);
}

// final heads: out / logits / gate (outputs concatenated per reference order)
__global__ __launch_bounds__(128) void head_kernel(
    const float* __restrict__ gph, const float* __restrict__ gfn,
    const float* __restrict__ gatep,
    const float* __restrict__ trw, const float* __restrict__ trb,
    const float* __restrict__ w1, const float* __restrict__ b1,
    const float* __restrict__ w2, const float* __restrict__ b2,
    float* __restrict__ dout){
  __shared__ float v64[64], outS[128], z1[128], lg[8];
  int b = blockIdx.x, tid = threadIdx.x;
  float gate = fminf(fmaxf(gatep[0], 0.f), 1.f);
  if (tid < 64) v64[tid] = gate * gph[b * 64 + tid] + (1.f - gate) * gfn[b * 64 + tid];
  __syncthreads();
  {
    float o = trb[tid];
    for (int f = 0; f < 64; ++f) o += v64[f] * trw[tid * 64 + f];
    o = fmaxf(o, 0.f);
    outS[tid] = o;
    dout[257 + b * H2 + tid] = o;          // 'out' section
  }
  __syncthreads();
  {
    float z = b1[tid];
    for (int j = 0; j < 128; ++j) z += outS[j] * w1[tid * 128 + j];
    z1[tid] = fmaxf(z, 0.f);
  }
  __syncthreads();
  if (tid < 8){
    float l = b2[tid];
    for (int h = 0; h < 128; ++h) l += z1[h] * w2[tid * 128 + h];
    lg[tid] = l;
  }
  __syncthreads();
  if (tid == 0){
    float mx = lg[0];
    for (int c = 1; c < 8; ++c) mx = fmaxf(mx, lg[c]);
    float se = 0.f;
    for (int c = 0; c < 8; ++c) se += expf(lg[c] - mx);
    float lse = logf(se);
    for (int c = 0; c < 8; ++c) dout[b * 8 + c] = lg[c] - mx - lse;  // logits
    if (b == 0) dout[256] = gate;                                    // gate scalar
  }
}

// ---------------------------------------------------------------------------
// Host launcher
// d_in order (setup_inputs insertion order, params recursively):
// 0 phy_x, 1 fun_x, 2 temp_embed, 3 spat_embed,
// 4..21  phy layers 0..2 x {w,b,rw,rb,g,be},
// 22..39 fun layers 0..2 x {w,b,rw,rb,g,be},
// 40..50 fusion {cw1,cb1,cw2,cb2,tw1,tb1,tw2,tb2,trw,trb,gate},
// 51..54 fc {w1,b1,w2,b2}
// ---------------------------------------------------------------------------
extern "C" void kernel_launch(void* const* d_in, const int* in_sizes, int n_in,
                              void* d_out, int out_size, void* d_ws, size_t ws_size,
                              hipStream_t stream){
  (void)in_sizes; (void)n_in; (void)out_size; (void)ws_size;
  const float* phy_x = (const float*)d_in[0];
  const float* fun_x = (const float*)d_in[1];
  const float* te    = (const float*)d_in[2];
  const float* se    = (const float*)d_in[3];
  struct LP { const float *w, *b, *rw, *rb, *g, *be; };
  LP Lp[2][3];
  for (int br = 0; br < 2; ++br)
    for (int l = 0; l < 3; ++l){
      int base = 4 + br * 18 + l * 6;
      Lp[br][l] = { (const float*)d_in[base+0], (const float*)d_in[base+1],
                    (const float*)d_in[base+2], (const float*)d_in[base+3],
                    (const float*)d_in[base+4], (const float*)d_in[base+5] };
    }
  const float* cw1 = (const float*)d_in[40]; const float* cb1 = (const float*)d_in[41];
  const float* cw2 = (const float*)d_in[42]; const float* cb2 = (const float*)d_in[43];
  const float* tw1 = (const float*)d_in[44]; const float* tb1 = (const float*)d_in[45];
  const float* tw2 = (const float*)d_in[46]; const float* tb2 = (const float*)d_in[47];
  const float* trw = (const float*)d_in[48]; const float* trb = (const float*)d_in[49];
  const float* gate= (const float*)d_in[50];
  const float* fw1 = (const float*)d_in[51]; const float* fb1 = (const float*)d_in[52];
  const float* fw2 = (const float*)d_in[53]; const float* fb2 = (const float*)d_in[54];
  float* dout = (float*)d_out;

  // workspace cursor
  char* ws = (char*)d_ws;
  size_t off = 0;
  auto alloc = [&](size_t bytes) -> char* {
    size_t o = (off + 255) & ~(size_t)255;
    off = o + bytes;
    return ws + o;
  };

  float* S    = (float*)alloc((size_t)Bn * Mm * Mm * 4);   // similarity (reused later)
  u16*   Ldyn = (u16*)  alloc((size_t)Bn * Mm * Mm * 2);
  u16*   Lphy = (u16*)  alloc((size_t)Mm * Mm * 2);
  u32*   bmk  = (u32*)  alloc((size_t)Rr * 24 * 4);
  float* dis  = (float*)alloc((size_t)Rr * 4);
  float* Hph  = (float*)alloc((size_t)Rr * Hh * 4);
  float* Hfn  = (float*)alloc((size_t)Rr * Hh * 4);
  u16*   Hbf  = (u16*)  alloc((size_t)Rr * Hh * 2);
  u16*   HbfT = (u16*)  alloc((size_t)Rr * Hh * 2);        // [B][C][M]
  u16*   Xnbf = (u16*)  alloc((size_t)Rr * Ff * 2);
  u16*   Wbf  = (u16*)  alloc((size_t)344064 * 2);
  float* mu   = (float*)alloc(256 * 4);
  float* rs   = (float*)alloc(256 * 4);
  float* alph = (float*)alloc((size_t)Rr * 4);
  float* gph  = (float*)alloc((size_t)Bn * 64 * 4);
  float* gfn  = (float*)alloc((size_t)Bn * 64 * 4);

  // recycle the S region for per-layer temporaries (S dead after ldyn_kernel)
  float* RES = S;                                            // Rr*256 f32
  float* Yb  = (float*)((char*)S + (size_t)Rr * Hh * 4);     // Rr*256 f32
  u16*   TX1 = (u16*)  ((char*)S + (size_t)Rr * Hh * 8);     // Rr*256 bf16

  const int cinA[3]  = {Ff, Hh, H2};
  const int coutA[3] = {Hh, H2, 64};
  int wOff[2][3], rwOff[2][3];
  {
    int o = 0;
    for (int br = 0; br < 2; ++br)
      for (int l = 0; l < 3; ++l){
        int ci = cinA[l], co = coutA[l];
        wOff[br][l]  = o; o += 2 * ci * co;   // w0T then w1T, each [co,ci]
        rwOff[br][l] = o; o += co * ci;       // rw already [co,ci]
      }
  }

  // ---- weights -> bf16 (w0/w1 transposed to [co,ci]; rw already [co,ci]) ----
  for (int br = 0; br < 2; ++br)
    for (int l = 0; l < 3; ++l){
      int ci = cinA[l], co = coutA[l];
      int nw = ci * co;
      cvtT_kernel<<<(nw + 255) / 256, 256, 0, stream>>>(Lp[br][l].w,      Wbf + wOff[br][l],      ci, co);
      cvtT_kernel<<<(nw + 255) / 256, 256, 0, stream>>>(Lp[br][l].w + nw, Wbf + wOff[br][l] + nw, ci, co);
      cvt_kernel <<<(nw + 255) / 256, 256, 0, stream>>>(Lp[br][l].rw,     Wbf + rwOff[br][l],     nw);
    }

  // ---- dynamic Laplacian ----
  xn_kernel<<<Rr, 64, 0, stream>>>(fun_x, Xnbf);
  gemm_bf16_kernel<<<dim3(Mm/64, Mm/64, Bn), 128, 0, stream>>>(
      Xnbf, (i64)Mm * Ff, Ff,
      Xnbf, (i64)Mm * Ff, Ff,
      nullptr, 0, nullptr, 0, 1,
      nullptr, S, nullptr, (i64)Mm * Mm, Mm, Ff);
  zdiag_kernel<<<Bn, Mm, 0, stream>>>(S);
  topk_kernel<<<(Rr + 255) / 256, 256, 0, stream>>>(S, bmk);
  deg_kernel<<<Rr, 64, 0, stream>>>(S, bmk, dis);
  ldyn_kernel<<<Rr, 64, 0, stream>>>(S, bmk, dis, Ldyn);
  lphy_kernel<<<Mm, 256, 0, stream>>>(Lphy);

  // ---- two branches x three ResidualChebConv layers ----
  for (int br = 0; br < 2; ++br){
    const float* x = br ? fun_x : phy_x;
    float* Hf = br ? Hfn : Hph;
    prep_kernel<<<(Rr * Ff) / 256, 256, 0, stream>>>(x, te, se, Hf, Hbf, HbfT);
    for (int l = 0; l < 3; ++l){
      int ci = cinA[l], co = coutA[l];
      const u16* w0T = Wbf + wOff[br][l];
      const u16* w1T = w0T + ci * co;
      const u16* rwb = Wbf + rwOff[br][l];
      // res = h @ rw^T + rb
      gemm_bf16_kernel<<<dim3(Rr/64, co/64, 1), 128, 0, stream>>>(
          Hbf, 0, ci,
          rwb, 0, ci,
          nullptr, 0, nullptr, 0, 1,
          Lp[br][l].rb, RES, nullptr, 0, co, ci);
      // tx1 = L @ h   (batched; physics L broadcast via stride 0; B = HbfT)
      const u16* Lb = br ? Ldyn : Lphy;
      gemm_bf16_kernel<<<dim3(Mm/64, ci/64, Bn), 128, 0, stream>>>(
          Lb, br ? (i64)Mm * Mm : 0, Mm,
          HbfT, (i64)ci * Mm, Mm,
          nullptr, 0, nullptr, 0, 1,
          nullptr, nullptr, TX1, (i64)Mm * ci, ci, Mm);
      // y = h @ w0 + tx1 @ w1 + b   (dual-source accumulate)
      gemm_bf16_kernel<<<dim3(Rr/64, co/64, 1), 128, 0, stream>>>(
          Hbf, 0, ci,
          w0T, 0, ci,
          TX1, ci, w1T, ci, 2,
          Lp[br][l].b, Yb, nullptr, 0, co, ci);
      bnstats_kernel<<<co, 256, 0, stream>>>(Yb, co, mu, rs);
      bnfin_kernel<<<(Rr * co) / 256, 256, 0, stream>>>(
          Yb, RES, Lp[br][l].g, Lp[br][l].be, mu, rs, Hf, Hbf, HbfT, co);
    }
  }

  // ---- fusion + heads ----
  alpha_kernel<<<Rr, 128, 0, stream>>>(Hph, Hfn, cw1, cb1, cw2, cb2, tw1, tb1, tw2, tb2, alph);
  gpool_kernel<<<Bn, 64, 0, stream>>>(alph, Hph, Hfn, gph, gfn);
  head_kernel<<<Bn, 128, 0, stream>>>(gph, gfn, gate, trw, trb, fw1, fb1, fw2, fb2, dout);
}
